// LSTM_63316407878387
// MI455X (gfx1250) — matrix operand order repaired and checked by reading, not compiled, not run on hardware
//
#include <hip/hip_runtime.h>
#include <stdint.h>
#include <math.h>

#define EMB   512
#define HID   1024
#define DIN   (HID + EMB)   // 1536
#define LSEQ  512
#define NB    64

typedef __attribute__((ext_vector_type(16))) __bf16    bf16x16;
typedef __attribute__((ext_vector_type(8)))  float     f32x8;
typedef __attribute__((ext_vector_type(4)))  uint32_t  u32x4;
typedef __attribute__((ext_vector_type(4)))  float     f32x4;

union Frag { u32x4 u[2]; bf16x16 v; };

__device__ __forceinline__ uint16_t f2bf(float f) {
  uint32_t u = __builtin_bit_cast(uint32_t, f);
  u += 0x7FFFu + ((u >> 16) & 1u);      // round-to-nearest-even
  return (uint16_t)(u >> 16);
}

// ---------------------------------------------------------------------------
// prep: WT[g*1024 + j][k] = bf16(W_g[k][j])   (gate-major, K contiguous)
// grid: 4096 blocks x 256
__global__ void lstm_prep_weights(const float* __restrict__ Wi, const float* __restrict__ Wf,
                                  const float* __restrict__ Wc, const float* __restrict__ Wo,
                                  uint16_t* __restrict__ WT) {
  int r = blockIdx.x;                 // r = g*1024 + j
  int g = r >> 10;
  int j = r & 1023;
  const float* W = (g == 0) ? Wi : (g == 1) ? Wf : (g == 2) ? Wc : Wo;
  uint16_t* dst = WT + (size_t)r * DIN;
  for (int k = threadIdx.x; k < DIN; k += 256)
    dst[k] = f2bf(W[(size_t)k * HID + j]);
}

// prep: XE[l][n][e] = bf16(emb[x[l][n]][e]).  grid: L*N blocks x 128
__global__ void lstm_prep_embed(const int* __restrict__ x, const float* __restrict__ emb,
                                uint16_t* __restrict__ XE) {
  int ln  = blockIdx.x;
  int tok = x[ln];
  const float* src = emb + (size_t)tok * EMB;
  uint16_t*    dst = XE  + (size_t)ln  * EMB;
  int e = threadIdx.x * 4;
  f32x4 f = *(const f32x4*)(src + e);
  dst[e + 0] = f2bf(f.x); dst[e + 1] = f2bf(f.y);
  dst[e + 2] = f2bf(f.z); dst[e + 3] = f2bf(f.w);
}

// prep: zero h double-buffer (65536 dwords) and c state (65536 floats)
__global__ void lstm_prep_zero(uint32_t* __restrict__ HB32, float* __restrict__ CS) {
  int i = blockIdx.x * 256 + threadIdx.x;     // grid 512 x 256 = 131072
  if (i < 65536) HB32[i] = 0u;
  else           CS[i - 65536] = 0.0f;
}

// ---------------------------------------------------------------------------
// One LSTM timestep. grid = (64 hidden blocks) x (2 batch halves), 128 thr.
// WG computes a 32(batch) x 64(=4 gates x 16 cols) slice of hx @ Wcat.
// Wave w (w=0..3) = gate w: one B fragment feeds TWO batch-row tiles.
__global__ void lstm_step(int t,
                          const uint16_t* __restrict__ WT,   // [4*HID][DIN] bf16
                          const uint16_t* __restrict__ XE,   // [L][NB][EMB] bf16
                          uint16_t*       __restrict__ HB,   // [2][NB][HID] bf16
                          float*          __restrict__ CS,   // [NB][HID] f32
                          const float* __restrict__ bi, const float* __restrict__ bf,
                          const float* __restrict__ bc, const float* __restrict__ bo,
                          float*          __restrict__ out)  // [L][NB][HID] f32
{
  __shared__ float gbuf[32 * 64];     // 8 KB: [batch_local][gate*16 + jloc]

  const int tid   = threadIdx.x;
  const int lane  = tid & 31;
  const int gate  = tid >> 5;          // wave id == gate (0=i,1=f,2=c~,3=o)
  const int l15   = lane & 15;
  const int khalf = lane >> 4;         // lane-half per ISA fragment layout

  const int jbase = blockIdx.x * 16;   // this WG's hidden-column block
  const int bhalf = blockIdx.y * 32;   // this WG's batch half

  // --- accumulators initialised with bias (C layout: N = lane&15 in every VGPR)
  const int n0 = gate * 16 + l15;      // n_local column of both accumulators
  const int jj = jbase + l15;
  const float bv = ((gate == 0) ? bi : (gate == 1) ? bf : (gate == 2) ? bc : bo)[jj];
  f32x8 acc0, acc1;                    // acc0: rows 0..15, acc1: rows 16..31 (local)
#pragma unroll
  for (int r = 0; r < 8; ++r) { acc0[r] = bv; acc1[r] = bv; }

  // --- operand row pointers
  const int m0 = bhalf + l15;          // global batch row for acc0's A fragment
  const int m1 = m0 + 16;              // global batch row for acc1's A fragment
  const uint16_t* brow  = WT + (size_t)(gate * HID + jj) * DIN;
  const uint16_t* hrow0 = HB + (size_t)(t & 1) * (NB * HID) + (size_t)m0 * HID;
  const uint16_t* hrow1 = HB + (size_t)(t & 1) * (NB * HID) + (size_t)m1 * HID;
  const uint16_t* xrow0 = XE + (size_t)t * (NB * EMB) + (size_t)m0 * EMB;
  const uint16_t* xrow1 = XE + (size_t)t * (NB * EMB) + (size_t)m1 * EMB;

  // --- K loop, recurrent part: hx[:, 0:1024) = h(t)
#pragma unroll 4
  for (int kk = 0; kk < HID; kk += 32) {
    Frag a0, a1, b;
    const uint16_t* bp = brow + kk + khalf * 8;           // lanes: K = kk+half*8..+7, kk+16+half*8..+7
    __builtin_prefetch(bp + 256, 0, 1);                    // global_prefetch_b8, 512B ahead
    b.u[0]  = *(const u32x4*)(bp);
    b.u[1]  = *(const u32x4*)(bp + 16);
    a0.u[0] = *(const u32x4*)(hrow0 + kk + khalf * 8);     // VGPR0-3: K = kk+half*8..+7
    a0.u[1] = *(const u32x4*)(hrow0 + kk + 16 + khalf * 8);
    a1.u[0] = *(const u32x4*)(hrow1 + kk + khalf * 8);
    a1.u[1] = *(const u32x4*)(hrow1 + kk + 16 + khalf * 8);
    acc0 = __builtin_amdgcn_wmma_f32_16x16x32_bf16(false, a0.v, false, b.v, (short)0, acc0, false, false);
    acc1 = __builtin_amdgcn_wmma_f32_16x16x32_bf16(false, a1.v, false, b.v, (short)0, acc1, false, false);
  }
  // --- K loop, input part: hx[:, 1024:1536) = xe[t]
#pragma unroll 4
  for (int kk = 0; kk < EMB; kk += 32) {
    Frag a0, a1, b;
    const uint16_t* bp = brow + HID + kk + khalf * 8;
    __builtin_prefetch(bp + 256, 0, 1);
    b.u[0]  = *(const u32x4*)(bp);
    b.u[1]  = *(const u32x4*)(bp + 16);
    a0.u[0] = *(const u32x4*)(xrow0 + kk + khalf * 8);
    a0.u[1] = *(const u32x4*)(xrow0 + kk + 16 + khalf * 8);
    a1.u[0] = *(const u32x4*)(xrow1 + kk + khalf * 8);
    a1.u[1] = *(const u32x4*)(xrow1 + kk + 16 + khalf * 8);
    acc0 = __builtin_amdgcn_wmma_f32_16x16x32_bf16(false, a0.v, false, b.v, (short)0, acc0, false, false);
    acc1 = __builtin_amdgcn_wmma_f32_16x16x32_bf16(false, a1.v, false, b.v, (short)0, acc1, false, false);
  }

  // --- spill accumulators (D layout: VGPR r -> local M = khalf*8 + r, N = lane&15)
  {
    const int mm = khalf * 8;
#pragma unroll
    for (int r = 0; r < 8; ++r) {
      gbuf[(mm + r) * 64 + n0]        = acc0[r];   // local rows 0..15
      gbuf[(mm + 16 + r) * 64 + n0]   = acc1[r];   // local rows 16..31
    }
  }
  __syncthreads();

  // --- elementwise cell update: 32 batches x 16 cols = 512 elems / 128 thr
  for (int e = tid; e < 32 * 16; e += 128) {
    const int b  = e >> 4;              // local batch row
    const int jl = e & 15;
    float iv = gbuf[b * 64 +      jl];
    float fv = gbuf[b * 64 + 16 + jl];
    float gv = gbuf[b * 64 + 32 + jl];
    float ov = gbuf[b * 64 + 48 + jl];
    iv = 1.0f / (1.0f + __expf(-iv));
    fv = 1.0f / (1.0f + __expf(-fv));
    ov = 1.0f / (1.0f + __expf(-ov));
    gv = tanhf(gv);
    const size_t idx = (size_t)(bhalf + b) * HID + jbase + jl;
    const float cn = fv * CS[idx] + iv * gv;
    CS[idx] = cn;
    const float hn = ov * tanhf(cn);
    out[(size_t)t * (NB * HID) + idx] = hn;
    HB[(size_t)((t + 1) & 1) * (NB * HID) + idx] = f2bf(hn);
  }
}

// ---------------------------------------------------------------------------
extern "C" void kernel_launch(void* const* d_in, const int* in_sizes, int n_in,
                              void* d_out, int out_size, void* d_ws, size_t ws_size,
                              hipStream_t stream) {
  (void)in_sizes; (void)n_in; (void)out_size; (void)ws_size;
  const int*   x   = (const int*)  d_in[0];
  const float* emb = (const float*)d_in[1];
  const float* Wi  = (const float*)d_in[2];
  const float* bi  = (const float*)d_in[3];
  const float* Wf  = (const float*)d_in[4];
  const float* bf  = (const float*)d_in[5];
  const float* Wc  = (const float*)d_in[6];
  const float* bc  = (const float*)d_in[7];
  const float* Wo  = (const float*)d_in[8];
  const float* bo  = (const float*)d_in[9];

  char* ws = (char*)d_ws;
  uint16_t* WT = (uint16_t*)(ws);              // 4*1024*1536*2   = 12,582,912 B
  uint16_t* XE = (uint16_t*)(ws + 12582912);   // 512*64*512*2    = 33,554,432 B
  uint16_t* HB = (uint16_t*)(ws + 46137344);   // 2*64*1024*2     =    262,144 B
  float*    CS = (float*)   (ws + 46399488);   // 64*1024*4       =    262,144 B

  lstm_prep_weights<<<4096, 256, 0, stream>>>(Wi, Wf, Wc, Wo, WT);
  lstm_prep_embed<<<LSEQ * NB, 128, 0, stream>>>(x, emb, XE);
  lstm_prep_zero<<<512, 256, 0, stream>>>((uint32_t*)HB, CS);

  float* out = (float*)d_out;
  for (int t = 0; t < LSEQ; ++t)
    lstm_step<<<dim3(HID / 16, 2), 128, 0, stream>>>(t, WT, XE, HB, CS, bi, bf, bc, bo, out);
}
